// SpikingNetwork_36292473651225
// MI455X (gfx1250) — matrix operand order repaired and checked by reading, not compiled, not run on hardware
//
#include <hip/hip_runtime.h>

// ---------------------------------------------------------------------------
// SNN (2-layer LIF, T=128) on MI455X gfx1250: ONE persistent bf16-WMMA kernel
// for the whole time scan (grid-barrier per step), membrane state in registers.
// B=128, T=128, I=1024, H1=H2=2048, O=512.
// ---------------------------------------------------------------------------

typedef __attribute__((ext_vector_type(16))) __bf16 v16bf;
typedef __attribute__((ext_vector_type(8)))  __bf16 v8bf;
typedef __attribute__((ext_vector_type(8)))  float  v8f;

union ABfrag { v16bf v; v8bf h[2]; };

#define BETA 0.9f
#define THR  1.0f
#define BB   128
#define TT   128
#define II   1024
#define HH   2048
#define OO   512

__device__ __forceinline__ v8f vzero() { v8f z = {}; return z; }

// Convert 8 consecutive f32 -> 8 bf16 (one 16B vector), with scale.
__device__ __forceinline__ v8bf cvt8(const float* __restrict__ p, float scale) {
  float4 f0 = *(const float4*)p;
  float4 f1 = *(const float4*)(p + 4);
  v8bf r;
  r[0] = (__bf16)(f0.x * scale); r[1] = (__bf16)(f0.y * scale);
  r[2] = (__bf16)(f0.z * scale); r[3] = (__bf16)(f0.w * scale);
  r[4] = (__bf16)(f1.x * scale); r[5] = (__bf16)(f1.y * scale);
  r[6] = (__bf16)(f1.z * scale); r[7] = (__bf16)(f1.w * scale);
  return r;
}

// Device-wide barrier: fresh counter per instance (zeroed by prep kernel each
// launch) -> no sense-reversal needed, deterministic under graph replay.
__device__ __forceinline__ void grid_barrier(int* __restrict__ ctr, int nwg) {
  __syncthreads();
  if (threadIdx.x == 0) {
    __hip_atomic_fetch_add(ctr, 1, __ATOMIC_ACQ_REL, __HIP_MEMORY_SCOPE_AGENT);
    while (__hip_atomic_load(ctr, __ATOMIC_ACQUIRE, __HIP_MEMORY_SCOPE_AGENT) < nwg) {
      __builtin_amdgcn_s_sleep(2);
    }
  }
  __syncthreads();
}

// One wave computes TWO 16x16 f32 tiles sharing the A fragment:
// C{a,b} += A(16xK from LDS) * B(Kx16), where B column n is row (n0+n) of
// row-major W[N][K]. Per-lane addressing follows the CDNA5 WMMA layout tables.
// `zoff` is an opaque zero: it keeps the weight-load addresses loop-VARIANT
// w.r.t. the timestep loop (no hoist -> no scratch spills) while preserving
// pointer provenance from the kernel argument (global_load, not flat_load).
__device__ __forceinline__ void wave_gemm2(const __bf16* As, int lda,
                                           const __bf16* W, int K,
                                           int n0a, int n0b, int lane,
                                           v8f& ca, v8f& cb) {
  const int r  = lane & 15;
  const int hi = lane >> 4;
  int zoff = 0;
  asm volatile("" : "+v"(zoff));             // opaque 0, fresh per invocation
  const __bf16* arow = As + r * lda + hi * 8;
  const __bf16* bap  = W + (size_t)(n0a + r) * K + hi * 8 + zoff;
  const __bf16* bbp  = W + (size_t)(n0b + r) * K + hi * 8 + zoff;
  ca = vzero(); cb = vzero();
#pragma unroll 4
  for (int k = 0; k < K; k += 32) {
    ABfrag a, ba, bb;
    a.h[0]  = *(const v8bf*)(arow + k);        // K = k+base    .. +7
    a.h[1]  = *(const v8bf*)(arow + k + 16);   // K = k+base+16 .. +23
    ba.h[0] = *(const v8bf*)(bap + k);
    ba.h[1] = *(const v8bf*)(bap + k + 16);
    bb.h[0] = *(const v8bf*)(bbp + k);
    bb.h[1] = *(const v8bf*)(bbp + k + 16);
    ca = __builtin_amdgcn_wmma_f32_16x16x32_bf16(false, a.v, false, ba.v,
                                                 (short)0, ca, false, false);
    cb = __builtin_amdgcn_wmma_f32_16x16x32_bf16(false, a.v, false, bb.v,
                                                 (short)0, cb, false, false);
  }
}

// ---- prep kernels ----------------------------------------------------------
__global__ void k_f32_to_bf16(const float* __restrict__ src, __bf16* __restrict__ dst, int n) {
  int i = blockIdx.x * blockDim.x + threadIdx.x;
  if (i < n) dst[i] = (__bf16)src[i];
}

__global__ void k_zero_i32(int* __restrict__ p, int n) {
  int i = blockIdx.x * blockDim.x + threadIdx.x;
  if (i < n) p[i] = 0;
}

// ---- persistent scan kernel -----------------------------------------------
// Grid (8,8): blockIdx.y = batch tile (16 rows), blockIdx.x = 256-column group.
// Each of 8 waves owns two 16-col tiles (n0a, n0b) used for BOTH layers, so
// v1/v2/acc for those tiles stay in registers across all 128 timesteps.
__global__ __launch_bounds__(256) void k_scan(const float*  __restrict__ x,
                                              const __bf16* __restrict__ W1bf,
                                              const float*  __restrict__ b1,
                                              const __bf16* __restrict__ W2bf,
                                              const float*  __restrict__ b2,
                                              const __bf16* __restrict__ Woutbf,
                                              const float*  __restrict__ bout,
                                              __bf16*       __restrict__ s1buf,   // 2 * BB*HH
                                              float*        __restrict__ accg,    // BB*HH
                                              float*        __restrict__ out,
                                              int*          __restrict__ barriers) {
  __shared__ __bf16 As[16 * HH];                 // 64 KB slab (layer1 uses 16*II)
  const int tid  = threadIdx.x;
  const int lane = tid & 31;
  const int wave = tid >> 5;
  const int nwg  = gridDim.x * gridDim.y;
  const int m0   = blockIdx.y * 16;
  const int n0a  = blockIdx.x * 256 + wave * 32;
  const int n0b  = n0a + 16;
  const int r    = lane & 15;
  const int mhi  = (lane >> 4) * 8;

  const float b1a = b1[n0a + r], b1b = b1[n0b + r];
  const float b2a = b2[n0a + r], b2b = b2[n0b + r];

  // Per-element recurrent state, resident in VGPRs for the whole scan.
  v8f v1a = vzero(), v1b = vzero();
  v8f v2a = vzero(), v2b = vzero();
  v8f aca = vzero(), acb = vzero();

  for (int t = 0; t < TT; ++t) {
    // -------- layer 1: stage x[.,t,:] (16 x II) as bf16 into LDS ----------
    __syncthreads();                              // As reads of prev iter done
    for (int ch = tid; ch < 16 * II / 8; ch += 256) {
      int row = ch >> 7;                          // II/8 = 128 chunks per row
      int col = (ch & 127) * 8;
      const float* p = x + ((size_t)(m0 + row) * TT + t) * II + col;
      *(v8bf*)&As[row * II + col] = cvt8(p, 1.0f);
    }
    __syncthreads();

    v8f ca, cb;
    wave_gemm2(As, II, W1bf, II, n0a, n0b, lane, ca, cb);

    // LIF layer 1 (reference semantics: reset keyed on input current) -> s1.
    __bf16* s1 = s1buf + (size_t)(t & 1) * BB * HH;
    for (int i = 0; i < 8; ++i) {
      int m = m0 + mhi + i;
      float cur, reset, mem;
      cur = ca[i] + b1a; reset = (cur > THR) ? THR : 0.0f;
      mem = BETA * cur + v1a[i] - reset; v1a[i] = mem;
      s1[(size_t)m * HH + n0a + r] = (__bf16)(((mem - THR) > 0.0f) ? 1.0f : 0.0f);
      cur = cb[i] + b1b; reset = (cur > THR) ? THR : 0.0f;
      mem = BETA * cur + v1b[i] - reset; v1b[i] = mem;
      s1[(size_t)m * HH + n0b + r] = (__bf16)(((mem - THR) > 0.0f) ? 1.0f : 0.0f);
    }

    grid_barrier(&barriers[t], nwg);              // s1 visible device-wide

    // -------- layer 2: stage s1 batch slab (16 x HH) into LDS -------------
    for (int ch = tid; ch < 16 * HH / 8; ch += 256) {
      int row = ch >> 8;                          // HH/8 = 256 chunks per row
      int col = (ch & 255) * 8;
      *(v8bf*)&As[row * HH + col] = *(const v8bf*)&s1[(size_t)(m0 + row) * HH + col];
    }
    __syncthreads();

    wave_gemm2(As, HH, W2bf, HH, n0a, n0b, lane, ca, cb);

    // LIF layer 2; accumulate output spikes in registers.
    for (int i = 0; i < 8; ++i) {
      float cur, reset, mem;
      cur = ca[i] + b2a; reset = (cur > THR) ? THR : 0.0f;
      mem = BETA * cur + v2a[i] - reset; v2a[i] = mem;
      aca[i] += ((mem - THR) > 0.0f) ? 1.0f : 0.0f;
      cur = cb[i] + b2b; reset = (cur > THR) ? THR : 0.0f;
      mem = BETA * cur + v2b[i] - reset; v2b[i] = mem;
      acb[i] += ((mem - THR) > 0.0f) ? 1.0f : 0.0f;
    }
  }

  // -------- publish spike counts, then fused readout GEMM ------------------
  for (int i = 0; i < 8; ++i) {
    int m = m0 + mhi + i;
    accg[(size_t)m * HH + n0a + r] = aca[i];
    accg[(size_t)m * HH + n0b + r] = acb[i];
  }
  grid_barrier(&barriers[TT], nwg);

  // out = (accg / T) @ Wout.T + bout.  Each WG: same m-tile, 64 output cols
  // (4 tiles, waves 0..3; waves 4..7 only help stage).
  const float invT = 1.0f / (float)TT;
  for (int ch = tid; ch < 16 * HH / 8; ch += 256) {
    int row = ch >> 8;
    int col = (ch & 255) * 8;
    const float* p = accg + (size_t)(m0 + row) * HH + col;
    *(v8bf*)&As[row * HH + col] = cvt8(p, invT);  // spike means k/128: exact bf16
  }
  __syncthreads();

  if (wave < 4) {
    const int n0r = blockIdx.x * 64 + wave * 16;  // 8 blocks * 64 = 512 cols
    const int hi  = lane >> 4;
    const __bf16* arow = As + r * HH + hi * 8;
    const __bf16* bp   = Woutbf + (size_t)(n0r + r) * HH + hi * 8;
    v8f c = vzero();
#pragma unroll 4
    for (int k = 0; k < HH; k += 32) {
      ABfrag a, b;
      a.h[0] = *(const v8bf*)(arow + k);
      a.h[1] = *(const v8bf*)(arow + k + 16);
      b.h[0] = *(const v8bf*)(bp + k);
      b.h[1] = *(const v8bf*)(bp + k + 16);
      c = __builtin_amdgcn_wmma_f32_16x16x32_bf16(false, a.v, false, b.v,
                                                  (short)0, c, false, false);
    }
    const float bias = bout[n0r + r];
    for (int i = 0; i < 8; ++i)
      out[(size_t)(m0 + mhi + i) * OO + n0r + r] = c[i] + bias;
  }
}

// ---------------------------------------------------------------------------
extern "C" void kernel_launch(void* const* d_in, const int* in_sizes, int n_in,
                              void* d_out, int out_size, void* d_ws, size_t ws_size,
                              hipStream_t stream) {
  (void)in_sizes; (void)n_in; (void)out_size; (void)ws_size;

  const float* x    = (const float*)d_in[0];   // (128,128,1024)
  const float* W1   = (const float*)d_in[1];   // (2048,1024)
  const float* b1   = (const float*)d_in[2];   // (2048,)
  const float* W2   = (const float*)d_in[3];   // (2048,2048)
  const float* b2   = (const float*)d_in[4];   // (2048,)
  const float* Wout = (const float*)d_in[5];   // (512,2048)
  const float* bout = (const float*)d_in[6];   // (512,)
  float* out = (float*)d_out;                  // (128,512) f32

  // Workspace layout (~16 MB).
  char* ws = (char*)d_ws;
  __bf16* W1bf     = (__bf16*)(ws);                        //  4 MB
  __bf16* W2bf     = (__bf16*)(ws + ((size_t)4  << 20));   //  8 MB
  __bf16* Woutbf   = (__bf16*)(ws + ((size_t)12 << 20));   //  2 MB
  __bf16* s1buf    = (__bf16*)(ws + ((size_t)14 << 20));   //  1 MB (double-buffered)
  float*  accg     = (float*) (ws + ((size_t)15 << 20));   //  1 MB
  int*    barriers = (int*)   (ws + ((size_t)16 << 20));   //  (TT+1) ints

  k_f32_to_bf16<<<(HH * II + 255) / 256, 256, 0, stream>>>(W1, W1bf, HH * II);
  k_f32_to_bf16<<<(HH * HH + 255) / 256, 256, 0, stream>>>(W2, W2bf, HH * HH);
  k_f32_to_bf16<<<(OO * HH + 255) / 256, 256, 0, stream>>>(Wout, Woutbf, OO * HH);
  k_zero_i32<<<1, 256, 0, stream>>>(barriers, TT + 1);

  // One persistent kernel for the entire scan + readout.
  // 64 WGs (8 batch tiles x 8 column groups) -> safely co-resident.
  k_scan<<<dim3(8, 8), 256, 0, stream>>>(x, W1bf, b1, W2bf, b2, Woutbf, bout,
                                         s1buf, accg, out, barriers);
}
